// LSTM_22823456211219
// MI455X (gfx1250) — matrix operand order repaired and checked by reading, not compiled, not run on hardware
//
#include <hip/hip_runtime.h>
#include <hip/hip_bf16.h>
#include <string.h>

// ---------------------------------------------------------------------------
// LSTM LM for MI455X (gfx1250): bf16 WMMA 16x16x32 with f32 accumulation.
//   B=16, T=512, H=512, L=2, V=32000
// Pipeline:
//   1) embed gather -> x[T][16][512] bf16
//   2) pack W/Wout into WMMA B-fragment layout (bf16)
//   3) per layer: Xproj = x @ Wx + b   (parallel WMMA GEMM, f32 out)
//                 recurrent kernel: 1 persistent block, h in LDS, c in regs
//   4) logits = h @ Wout + bout        (parallel WMMA GEMM, writes d_out)
// All tile counts divide block tiles exactly -> no guards -> no EXEC masking
// around WMMA.  K-loops are double-buffered to hide load latency.
// ---------------------------------------------------------------------------

typedef __bf16 bf16_t;
typedef __attribute__((ext_vector_type(16))) __bf16 v16bf;
typedef __attribute__((ext_vector_type(8)))  float  v8f;

union FragU { v16bf v; uint4 q[2]; };

#define TT   512
#define BB   16
#define HH   512
#define VV   32000
#define G4H  2048   // 4*H

// ---------------------------------------------------------------------------
__device__ __forceinline__ v8f wmma_bf16(v16bf a, v16bf b, v8f c) {
  // (neg_a, A, neg_b, B, c_mod, C, reuse_a, reuse_b)
  return __builtin_amdgcn_wmma_f32_16x16x32_bf16(false, a, false, b,
                                                 (short)0, c, false, false);
}

// A fragment (16x32 bf16, MxK) from row-major [rows][lda] bf16.
// Lane layout: row = lane&15; q0 holds K = k0+hi .. +7, q1 holds K = k0+16+hi..+7
// where hi = 8*(lane>=16).  Matches ISA 16-bit A 16x32 layout.
__device__ __forceinline__ v16bf load_a_frag(const bf16_t* base, int lda,
                                             int k0, int lane) {
  const int row = lane & 15;
  const int hi  = (lane >> 4) << 3;       // 0 or 8
  FragU u;
  u.q[0] = *(const uint4*)(base + (size_t)row * lda + k0 + hi);
  u.q[1] = *(const uint4*)(base + (size_t)row * lda + k0 + 16 + hi);
  return u.v;
}

// B fragment (32x16 bf16, KxN) from pre-packed fragments: 512 bf16 per frag,
// lane reads its 32 contiguous bytes (elements lane*16 .. lane*16+15).
__device__ __forceinline__ v16bf load_b_frag(const bf16_t* packed,
                                             size_t fragIdx, int lane) {
  const bf16_t* p = packed + fragIdx * 512 + (size_t)lane * 16;
  FragU u;
  u.q[0] = ((const uint4*)p)[0];
  u.q[1] = ((const uint4*)p)[1];
  return u.v;
}

__device__ __forceinline__ float sigf(float x) {
  return 1.0f / (1.0f + __expf(-x));
}

// ---------------------------------------------------------------------------
// 1) Embedding gather -> bf16 activations, layout [t][b][h]
__global__ __launch_bounds__(256)
void embed_kernel(const int* __restrict__ tokens, const float* __restrict__ embed,
                  bf16_t* __restrict__ x) {
  int idx = blockIdx.x * 256 + threadIdx.x;            // over T*B*H
  if (idx >= TT * BB * HH) return;
  int t  = idx >> 13;                                  // /(B*H)=8192
  int r  = idx & 8191;
  int bb = r >> 9;                                     // /H
  int h  = r & (HH - 1);
  int tok = tokens[bb * TT + t];
  x[idx] = (bf16_t)embed[(size_t)tok * HH + h];
}

// ---------------------------------------------------------------------------
// 2) Pack f32 weight matrix [rows][srcLd] into bf16 WMMA B fragments.
// dst element (frag, lane, j) = src[rowOff + kT*32 + (j&7) + 16*(j>>3) + 8*(lane>=16)][nT*16 + (lane&15)]
// with frag = nT*Kfrags + kT (k-contiguous for streaming in the k-loop).
__global__ __launch_bounds__(256)
void pack_w_kernel(const float* __restrict__ src, bf16_t* __restrict__ dst,
                   int rowOff, int srcLd, int total, int Kfrags) {
  int idx = blockIdx.x * 256 + threadIdx.x;
  if (idx >= total) return;
  int j    = idx & 15;
  int lane = (idx >> 4) & 31;
  int frag = idx >> 9;
  int k = (frag % Kfrags) * 32 + (j & 7) + ((j >> 3) << 4) + ((lane >> 4) << 3);
  int n = (frag / Kfrags) * 16 + (lane & 15);
  dst[idx] = (bf16_t)src[(size_t)(rowOff + k) * srcLd + n];
}

// ---------------------------------------------------------------------------
// 3) Tiled bf16 WMMA GEMM:  C[m][n] = A[m][k] @ Bpacked + bias[n]   (f32 out)
//    M = 8192 (Mtile = blockIdx.y = one timestep, rows = batch)
//    block = 256 threads = 8 waves; wave owns NPW n-tiles (no tail: Ntiles
//    must equal gridDim.x * 8 * NPW).  Double-buffered k-loop.
//    remapT: 0 -> row m stored directly; else out row = (m%16)*remapT + m/16
//    (maps [t][b] activation rows to [b][t] logits rows).
template <int NPW>
__global__ __launch_bounds__(256)
void gemm_bf16_kernel(const bf16_t* __restrict__ A, const bf16_t* __restrict__ Bp,
                      const float* __restrict__ bias, float* __restrict__ C,
                      int Kfrags, int ldC, int remapT) {
  const int lane  = threadIdx.x & 31;
  const int wv    = threadIdx.x >> 5;
  const int mtile = blockIdx.y;
  const int nt0   = (blockIdx.x * 8 + wv) * NPW;
  const int lda   = Kfrags * 32;
  const bf16_t* Arow = A + (size_t)mtile * 16 * lda;
  const size_t fbase = (size_t)nt0 * Kfrags;           // first n-tile's frag base

  v8f acc[NPW] = {};

  // prologue: fragments for k = 0
  v16bf a = load_a_frag(Arow, lda, 0, lane);
  v16bf bc[NPW];
#pragma unroll
  for (int i = 0; i < NPW; ++i)
    bc[i] = load_b_frag(Bp, fbase + (size_t)i * Kfrags, lane);

  for (int k = 0; k < Kfrags - 1; ++k) {
    __builtin_prefetch((const void*)(Bp + (fbase + k + 2) * 512), 0, 3);
    v16bf an = load_a_frag(Arow, lda, (k + 1) * 32, lane);
    v16bf bn[NPW];
#pragma unroll
    for (int i = 0; i < NPW; ++i)
      bn[i] = load_b_frag(Bp, fbase + (size_t)i * Kfrags + k + 1, lane);
#pragma unroll
    for (int i = 0; i < NPW; ++i) acc[i] = wmma_bf16(a, bc[i], acc[i]);
    a = an;
#pragma unroll
    for (int i = 0; i < NPW; ++i) bc[i] = bn[i];
  }
#pragma unroll
  for (int i = 0; i < NPW; ++i) acc[i] = wmma_bf16(a, bc[i], acc[i]);

  // epilogue: D frag layout -> M = r + 8*(lane>=16), N = lane&15
  const int Mo = (lane >> 4) << 3;
#pragma unroll
  for (int i = 0; i < NPW; ++i) {
    int col  = (nt0 + i) * 16 + (lane & 15);
    float bs = bias[col];
#pragma unroll
    for (int r = 0; r < 8; ++r) {
      int gRow = mtile * 16 + Mo + r;                  // = t*16 + b
      size_t orow = remapT ? (size_t)(gRow & 15) * remapT + (gRow >> 4)
                           : (size_t)gRow;
      C[orow * (size_t)ldC + col] = acc[i][r] + bs;
    }
  }
}

// ---------------------------------------------------------------------------
// 4) Persistent recurrent kernel: 1 block, 1024 threads = 32 waves.
//    Per step: gates = Xproj[t] (x@Wx+b precomputed) + h @ Wh.
//    Wave w owns hidden columns [16w,16w+16): its 4 n-tiles are the i/f/g/o
//    columns for those units -> the c/h update is wave-local.  h (bf16) lives
//    in LDS (A-matrix for next step via ds_load_b128); c lives in a v8f reg.
__global__ __launch_bounds__(1024)
void lstm_recurrent_kernel(const float* __restrict__ Xproj,   // [T][16][2048]
                           const bf16_t* __restrict__ Whp,    // packed 128x16 frags
                           bf16_t* __restrict__ Hout) {       // [T][16][512]
  __shared__ bf16_t hlds[BB * HH];
  const int tid  = threadIdx.x;
  const int lane = tid & 31;
  const int w    = tid >> 5;                  // 0..31
  const int colL = (w << 4) + (lane & 15);    // hidden column owned by this lane
  const int Mo   = (lane >> 4) << 3;          // frag row offset: 0 or 8

  for (int i = tid; i < BB * HH; i += 1024) hlds[i] = (bf16_t)0.0f;
  v8f c = {};
  __syncthreads();

  for (int t = 0; t < TT; ++t) {
    const float* xp = Xproj + (size_t)t * BB * G4H;
    // init accumulators with precomputed x-projection (bias already folded in)
    v8f acc[4] = {};
#pragma unroll
    for (int r = 0; r < 8; ++r) {
      const float* row = xp + (size_t)(Mo + r) * G4H;
#pragma unroll
      for (int g = 0; g < 4; ++g) acc[g][r] = row[g * 512 + colL];
    }
    // recurrent matmul: h[16x512] @ Wh[512x2048], 16 k-frags, double-buffered
    v16bf a = load_a_frag(hlds, HH, 0, lane);            // ds_load_b128
    v16bf bc[4];
#pragma unroll
    for (int g = 0; g < 4; ++g)
      bc[g] = load_b_frag(Whp, (size_t)(w + 32 * g) * 16, lane);
    for (int k = 0; k < 15; ++k) {
      v16bf an = load_a_frag(hlds, HH, (k + 1) * 32, lane);
      v16bf bn[4];
#pragma unroll
      for (int g = 0; g < 4; ++g)
        bn[g] = load_b_frag(Whp, (size_t)(w + 32 * g) * 16 + k + 1, lane);
#pragma unroll
      for (int g = 0; g < 4; ++g) acc[g] = wmma_bf16(a, bc[g], acc[g]);
      a = an;
#pragma unroll
      for (int g = 0; g < 4; ++g) bc[g] = bn[g];
    }
#pragma unroll
    for (int g = 0; g < 4; ++g) acc[g] = wmma_bf16(a, bc[g], acc[g]);

    // gate nonlinearity + state update (wave-local)
    v8f hnew;
#pragma unroll
    for (int r = 0; r < 8; ++r) {
      float ig = sigf(acc[0][r]);
      float fg = sigf(acc[1][r]);
      float gg = tanhf(acc[2][r]);
      float og = sigf(acc[3][r]);
      float cn = fg * c[r] + ig * gg;
      c[r]     = cn;
      hnew[r]  = og * tanhf(cn);
    }
    __syncthreads();   // all waves done reading h(t-1) from LDS
#pragma unroll
    for (int r = 0; r < 8; ++r) {
      bf16_t hb = (bf16_t)hnew[r];
      hlds[(Mo + r) * HH + colL] = hb;
      Hout[(size_t)t * BB * HH + (size_t)(Mo + r) * HH + colL] = hb;
    }
    __syncthreads();   // h(t) visible before next step's A loads
  }
}

// ---------------------------------------------------------------------------
extern "C" void kernel_launch(void* const* d_in, const int* in_sizes, int n_in,
                              void* d_out, int out_size, void* d_ws, size_t ws_size,
                              hipStream_t stream) {
  const int*   tokens = (const int*)d_in[0];     // [B,T]
  const float* embedW = (const float*)d_in[1];   // [V,H]
  const float* W      = (const float*)d_in[2];   // [L,2H,4H]
  const float* bvec   = (const float*)d_in[3];   // [L,4H]
  const float* Wout   = (const float*)d_in[4];   // [H,V]
  const float* bout   = (const float*)d_in[5];   // [V]
  float* out = (float*)d_out;                    // [B,T,V]
  char*  ws  = (char*)d_ws;

  // workspace layout (all offsets 256B aligned)
  bf16_t* xa    = (bf16_t*)(ws);                       //  8,388,608 B  x ping
  bf16_t* xb    = (bf16_t*)(ws + 8388608);             //  8,388,608 B  x pong
  float*  xproj = (float*) (ws + 16777216);            // 67,108,864 B  [T][16][2048]
  bf16_t* pWx   = (bf16_t*)(ws + 83886080);            //  4,194,304 B  2 layers
  bf16_t* pWh   = (bf16_t*)(ws + 88080384);            //  4,194,304 B  2 layers
  bf16_t* pWo   = (bf16_t*)(ws + 92274688);            // 32,768,000 B
  const size_t wLayerStride  = (size_t)1024 * 2048;    // f32 elems per layer of W
  const size_t pLayerStride  = (size_t)128 * 16 * 512; // packed bf16 elems per layer

  // 1) embedding gather
  embed_kernel<<<(TT * BB * HH) / 256, 256, 0, stream>>>(tokens, embedW, xa);

  // 2) pack weights: Wx (rows 0..511), Wh (rows 512..1023), Wout
  const int wxTotal = 128 * 16 * 512;                  // per layer
  for (int l = 0; l < 2; ++l) {
    pack_w_kernel<<<wxTotal / 256, 256, 0, stream>>>(
        W + l * wLayerStride, pWx + l * pLayerStride, 0,   G4H, wxTotal, 16);
    pack_w_kernel<<<wxTotal / 256, 256, 0, stream>>>(
        W + l * wLayerStride, pWh + l * pLayerStride, 512, G4H, wxTotal, 16);
  }
  const int woTotal = 2000 * 16 * 512;
  pack_w_kernel<<<(woTotal + 255) / 256, 256, 0, stream>>>(
      Wout, pWo, 0, VV, woTotal, 16);

  // 3) LSTM layers
  bf16_t* xin  = xa;
  bf16_t* xout = xb;
  for (int l = 0; l < 2; ++l) {
    // Xproj = x @ Wx + b : Ntiles=128 = 4 blocks * 8 waves * NPW(4)
    dim3 gx(4, 512);
    gemm_bf16_kernel<4><<<gx, 256, 0, stream>>>(
        xin, pWx + l * pLayerStride, bvec + (size_t)l * G4H, xproj,
        /*Kfrags=*/16, /*ldC=*/G4H, /*remapT=*/0);
    // recurrent scan (serial over T, single persistent workgroup)
    lstm_recurrent_kernel<<<1, 1024, 0, stream>>>(
        xproj, pWh + l * pLayerStride, xout);
    bf16_t* tmp = xin; xin = xout; xout = tmp;
  }

  // 4) logits = h @ Wout + bout : Ntiles=2000 = 50 blocks * 8 waves * NPW(5)
  //    remap rows [t][b] -> [b][t]
  dim3 gl(50, 512);
  gemm_bf16_kernel<5><<<gl, 256, 0, stream>>>(
      xin, pWo, bout, out,
      /*Kfrags=*/16, /*ldC=*/VV, /*remapT=*/TT);
}
